// Conv1d_21148418966219
// MI455X (gfx1250) — matrix operand order repaired and checked by reading, not compiled, not run on hardware
//
#include <hip/hip_runtime.h>

// Conv1d [32,128,4096] * [256,128,9] pad=4 -> [32,256,4096], fp32.
// bf16x3 fp32-emulation on the WMMA pipe (3x v_wmma_f32_16x16x32_bf16 per
// K=32 chunk, fp32 accumulate). Fast path: one-time pre-pass splits fp32 into
// bf16 hi/lo planes in d_ws (transposed + halo-padded), main kernel stages
// tiles with async global->LDS (ASYNCcnt) and double-buffers the weight tile.

typedef __attribute__((ext_vector_type(16))) __bf16 v16bf;
typedef __attribute__((ext_vector_type(8)))  float  v8f;

#define C_IN   128
#define W_IN   4096
#define C_OUT  256
#define KSZ    9
#define PAD    4
#define BATCH  32

#define MTILE  128                 // c_out per workgroup
#define NTILE  128                 // w positions per workgroup
#define CCHUNK 32                  // WMMA bf16 K
#define NCHUNK (C_IN / CCHUNK)     // 4
#define NSTAGE (NTILE + KSZ - 1)   // 136 staged w rows (halo = K-1)
#define ROWP   40                  // padded LDS row: 32 data + 8 pad ushorts
#define WPADD  (W_IN + 2 * PAD)    // 4104 padded input rows in ws

// ws layout (ushort elements):
//  wIhi: [B][NCHUNK][WPADD][32]   (transposed input, zero halo rows baked in)
//  wIlo: same
//  wWhi: [KSZ][NCHUNK][C_OUT][32] (weight slices per tap/chunk)
//  wWlo: same
#define SI_ELEMS ((size_t)BATCH * NCHUNK * WPADD * 32)
#define SW_ELEMS ((size_t)KSZ * NCHUNK * C_OUT * 32)

union Frag16 { uint4 q[2]; v16bf v; };

__device__ __forceinline__ void bf16split(float x, unsigned short& hi, unsigned short& lo) {
  unsigned xu = __float_as_uint(x);
  hi = (unsigned short)(xu >> 16);
  float lof = x - __uint_as_float((unsigned)hi << 16);
  lo = (unsigned short)(__float_as_uint(lof) >> 16);
}

__device__ __forceinline__ void async_b128(unsigned lds_off, unsigned gl_off, const void* base) {
  asm volatile("global_load_async_to_lds_b128 %0, %1, %2"
               :: "v"(lds_off), "v"(gl_off), "s"(base) : "memory");
}
__device__ __forceinline__ void wait_async0() {
  asm volatile("s_wait_asynccnt 0" ::: "memory");
}
__device__ __forceinline__ unsigned lds_addr(const void* p) {
  return (unsigned)(size_t)p;   // low 32 bits of flat shared addr = LDS offset
}

// ---------------- Pre-pass 1: input split+transpose+pad ----------------
// grid (17, NCHUNK, BATCH), 256 thr. Tile: 32 c x 256 padded rows.
__global__ __launch_bounds__(256)
void prep_input(const float* __restrict__ in, unsigned short* __restrict__ dhi,
                unsigned short* __restrict__ dlo) {
  __shared__ unsigned tile[32][264];   // hi<<16 | lo
  const int tid = threadIdx.x;
  const int r0  = blockIdx.x * 256;
  const int ch  = blockIdx.y;
  const int b   = blockIdx.z;

  for (int i = tid; i < 32 * 256; i += 256) {
    int c  = i >> 8;
    int wp = i & 255;
    int w  = r0 + wp - PAD;
    float x = 0.0f;
    if (w >= 0 && w < W_IN)
      x = in[((size_t)b * C_IN + ch * CCHUNK + c) * (size_t)W_IN + w];
    unsigned short hi, lo;
    bf16split(x, hi, lo);
    tile[c][wp] = ((unsigned)hi << 16) | lo;
  }
  __syncthreads();

  int row = r0 + tid;
  if (row < WPADD) {
    union { unsigned short s[32]; uint4 q[4]; } vh, vl;
#pragma unroll
    for (int c = 0; c < 32; ++c) {
      unsigned p = tile[c][tid];
      vh.s[c] = (unsigned short)(p >> 16);
      vl.s[c] = (unsigned short)(p & 0xffff);
    }
    size_t base = (((size_t)b * NCHUNK + ch) * WPADD + row) * 32;
    uint4* ph = (uint4*)(dhi + base);
    uint4* pl = (uint4*)(dlo + base);
#pragma unroll
    for (int q = 0; q < 4; ++q) { ph[q] = vh.q[q]; pl[q] = vl.q[q]; }
  }
}

// ---------------- Pre-pass 2: weight split+repack ----------------
// dst layout [k][chunk][co][cc]
__global__ __launch_bounds__(256)
void prep_weight(const float* __restrict__ wt, unsigned short* __restrict__ dhi,
                 unsigned short* __restrict__ dlo) {
  int idx = blockIdx.x * 256 + threadIdx.x;      // < KSZ*NCHUNK*C_OUT*32
  if (idx >= (int)SW_ELEMS) return;
  int cc = idx & 31;
  int co = (idx >> 5) & (C_OUT - 1);
  int ch = (idx >> 13) & (NCHUNK - 1);
  int k  = idx >> 15;
  float x = wt[((size_t)co * C_IN + ch * CCHUNK + cc) * KSZ + k];
  unsigned short hi, lo;
  bf16split(x, hi, lo);
  dhi[idx] = hi;
  dlo[idx] = lo;
}

// ---------------- Fast main kernel: async staging, dbl-buffered weights ----
__global__ __launch_bounds__(256)
void conv1d_wmma_bf16x3_async(const unsigned short* __restrict__ wIhi,
                              const unsigned short* __restrict__ wIlo,
                              const unsigned short* __restrict__ wWhi,
                              const unsigned short* __restrict__ wWlo,
                              const float* __restrict__ bias,
                              float* __restrict__ out) {
  __shared__ __attribute__((aligned(16))) unsigned short sWhi[2][MTILE][ROWP];
  __shared__ __attribute__((aligned(16))) unsigned short sWlo[2][MTILE][ROWP];
  __shared__ __attribute__((aligned(16))) unsigned short sIhi[NSTAGE][ROWP];
  __shared__ __attribute__((aligned(16))) unsigned short sIlo[NSTAGE][ROWP];

  const int tid    = threadIdx.x;
  const int lane   = tid & 31;
  const int wave   = tid >> 5;
  const int waveM  = wave & 3;
  const int waveN  = wave >> 2;
  const int lm     = lane & 15;
  const bool laneHi = lane >= 16;

  const int w0  = blockIdx.x * NTILE;
  const int co0 = blockIdx.y * MTILE;
  const int b   = blockIdx.z;

  v8f acc[2][4] = {};
  int buf = 0;

  for (int chk = 0; chk < NCHUNK; ++chk) {
    __syncthreads();   // prior compute done before overwriting sI
    // async-stage input tile (136 rows x 64B from pre-transposed planes)
    {
      unsigned gsl = (unsigned)((((b * NCHUNK + chk) * WPADD) + w0) * 32) * 2u;
      for (int i = tid; i < NSTAGE * 4; i += 256) {
        int r = i >> 2, q = i & 3;
        async_b128(lds_addr(&sIhi[r][0]) + q * 16, gsl + r * 64 + q * 16, wIhi);
      }
      for (int i = tid; i < NSTAGE * 4; i += 256) {
        int r = i >> 2, q = i & 3;
        async_b128(lds_addr(&sIlo[r][0]) + q * 16, gsl + r * 64 + q * 16, wIlo);
      }
    }
    // async-stage weight tap 0 into sW[buf]
    {
      unsigned gsl = (unsigned)((((0 * NCHUNK + chk) * C_OUT) + co0) * 32) * 2u;
      for (int i = tid; i < MTILE * 4; i += 256) {
        int r = i >> 2, q = i & 3;
        async_b128(lds_addr(&sWhi[buf][r][0]) + q * 16, gsl + r * 64 + q * 16, wWhi);
      }
      for (int i = tid; i < MTILE * 4; i += 256) {
        int r = i >> 2, q = i & 3;
        async_b128(lds_addr(&sWlo[buf][r][0]) + q * 16, gsl + r * 64 + q * 16, wWlo);
      }
    }
    wait_async0();
    __syncthreads();

    for (int kk = 0; kk < KSZ; ++kk) {
      // prefetch next tap into the other buffer while computing this one
      if (kk + 1 < KSZ) {
        unsigned gsl = (unsigned)(((((kk + 1) * NCHUNK + chk) * C_OUT) + co0) * 32) * 2u;
        for (int i = tid; i < MTILE * 4; i += 256) {
          int r = i >> 2, q = i & 3;
          async_b128(lds_addr(&sWhi[buf ^ 1][r][0]) + q * 16, gsl + r * 64 + q * 16, wWhi);
        }
        for (int i = tid; i < MTILE * 4; i += 256) {
          int r = i >> 2, q = i & 3;
          async_b128(lds_addr(&sWlo[buf ^ 1][r][0]) + q * 16, gsl + r * 64 + q * 16, wWlo);
        }
      }

      // A fragments: 16x32 bf16, lane layout per ISA 7.12.2
      Frag16 ahi[2], alo[2];
      const int ao = laneHi ? 8 : 0;
#pragma unroll
      for (int mt = 0; mt < 2; ++mt) {
        const unsigned short* rh = &sWhi[buf][waveM * 32 + mt * 16 + lm][0];
        const unsigned short* rl = &sWlo[buf][waveM * 32 + mt * 16 + lm][0];
        ahi[mt].q[0] = *(const uint4*)(rh + ao);
        ahi[mt].q[1] = *(const uint4*)(rh + 16 + ao);
        alo[mt].q[0] = *(const uint4*)(rl + ao);
        alo[mt].q[1] = *(const uint4*)(rl + 16 + ao);
      }
      // B fragments: 32x16 bf16; tap shift = +kk offset into halo tile
      Frag16 bhi[4], blo[4];
      const int bo = laneHi ? 8 : 0;
#pragma unroll
      for (int nt = 0; nt < 4; ++nt) {
        int wl = waveN * 64 + nt * 16 + lm + kk;
        const unsigned short* rh = &sIhi[wl][0];
        const unsigned short* rl = &sIlo[wl][0];
        bhi[nt].q[0] = *(const uint4*)(rh + bo);
        bhi[nt].q[1] = *(const uint4*)(rh + bo + 16);
        blo[nt].q[0] = *(const uint4*)(rl + bo);
        blo[nt].q[1] = *(const uint4*)(rl + bo + 16);
      }

#pragma unroll
      for (int mt = 0; mt < 2; ++mt) {
#pragma unroll
        for (int nt = 0; nt < 4; ++nt) {
          acc[mt][nt] = __builtin_amdgcn_wmma_f32_16x16x32_bf16(
              false, ahi[mt].v, false, bhi[nt].v, (short)0, acc[mt][nt], false, false);
          acc[mt][nt] = __builtin_amdgcn_wmma_f32_16x16x32_bf16(
              false, ahi[mt].v, false, blo[nt].v, (short)0, acc[mt][nt], false, false);
          acc[mt][nt] = __builtin_amdgcn_wmma_f32_16x16x32_bf16(
              false, alo[mt].v, false, bhi[nt].v, (short)0, acc[mt][nt], false, false);
        }
      }

      wait_async0();     // my prefetch writes for next buffer are in LDS
      __syncthreads();   // everyone done reading sW[buf] / writing sW[buf^1]
      buf ^= 1;
    }
  }

#pragma unroll
  for (int mt = 0; mt < 2; ++mt) {
#pragma unroll
    for (int nt = 0; nt < 4; ++nt) {
#pragma unroll
      for (int r = 0; r < 8; ++r) {
        int co = co0 + waveM * 32 + mt * 16 + r + (laneHi ? 8 : 0);
        int w  = w0 + waveN * 64 + nt * 16 + lm;
        out[((size_t)b * C_OUT + co) * (size_t)W_IN + w] = acc[mt][nt][r] + bias[co];
      }
    }
  }
}

// ---------------- Fallback (self-contained, no ws needed) ----------------
__global__ __launch_bounds__(256)
void conv1d_wmma_bf16x3_fb(const float* __restrict__ in,
                           const float* __restrict__ wt,
                           const float* __restrict__ bias,
                           float* __restrict__ out) {
  __shared__ __attribute__((aligned(16))) unsigned short sWhi[MTILE][ROWP];
  __shared__ __attribute__((aligned(16))) unsigned short sWlo[MTILE][ROWP];
  __shared__ __attribute__((aligned(16))) unsigned short sIhi[NSTAGE][ROWP];
  __shared__ __attribute__((aligned(16))) unsigned short sIlo[NSTAGE][ROWP];

  const int tid    = threadIdx.x;
  const int lane   = tid & 31;
  const int wave   = tid >> 5;
  const int waveM  = wave & 3;
  const int waveN  = wave >> 2;
  const int lm     = lane & 15;
  const bool laneHi = lane >= 16;

  const int w0  = blockIdx.x * NTILE;
  const int co0 = blockIdx.y * MTILE;
  const int b   = blockIdx.z;

  v8f acc[2][4] = {};

  for (int c0 = 0; c0 < C_IN; c0 += CCHUNK) {
    __syncthreads();
    for (int idx = tid; idx < CCHUNK * NSTAGE; idx += 256) {
      int c = idx / NSTAGE;
      int w = idx - c * NSTAGE;
      int wg = w0 - PAD + w;
      float x = 0.0f;
      if (wg >= 0 && wg < W_IN)
        x = in[((size_t)b * C_IN + (c0 + c)) * (size_t)W_IN + wg];
      unsigned short hi, lo;
      bf16split(x, hi, lo);
      sIhi[w][c] = hi;
      sIlo[w][c] = lo;
    }
    for (int kk = 0; kk < KSZ; ++kk) {
      __syncthreads();
      for (int idx = tid; idx < MTILE * CCHUNK; idx += 256) {
        int co = idx >> 5;
        int c  = idx & 31;
        float x = wt[(((size_t)(co0 + co)) * C_IN + (c0 + c)) * KSZ + kk];
        unsigned short hi, lo;
        bf16split(x, hi, lo);
        sWhi[co][c] = hi;
        sWlo[co][c] = lo;
      }
      __syncthreads();

      Frag16 ahi[2], alo[2];
      const int ao = laneHi ? 8 : 0;
#pragma unroll
      for (int mt = 0; mt < 2; ++mt) {
        const unsigned short* rh = &sWhi[waveM * 32 + mt * 16 + lm][0];
        const unsigned short* rl = &sWlo[waveM * 32 + mt * 16 + lm][0];
        ahi[mt].q[0] = *(const uint4*)(rh + ao);
        ahi[mt].q[1] = *(const uint4*)(rh + 16 + ao);
        alo[mt].q[0] = *(const uint4*)(rl + ao);
        alo[mt].q[1] = *(const uint4*)(rl + 16 + ao);
      }
      Frag16 bhi[4], blo[4];
      const int bo = laneHi ? 8 : 0;
#pragma unroll
      for (int nt = 0; nt < 4; ++nt) {
        int wl = waveN * 64 + nt * 16 + lm + kk;
        const unsigned short* rh = &sIhi[wl][0];
        const unsigned short* rl = &sIlo[wl][0];
        bhi[nt].q[0] = *(const uint4*)(rh + bo);
        bhi[nt].q[1] = *(const uint4*)(rh + bo + 16);
        blo[nt].q[0] = *(const uint4*)(rl + bo);
        blo[nt].q[1] = *(const uint4*)(rl + bo + 16);
      }
#pragma unroll
      for (int mt = 0; mt < 2; ++mt) {
#pragma unroll
        for (int nt = 0; nt < 4; ++nt) {
          acc[mt][nt] = __builtin_amdgcn_wmma_f32_16x16x32_bf16(
              false, ahi[mt].v, false, bhi[nt].v, (short)0, acc[mt][nt], false, false);
          acc[mt][nt] = __builtin_amdgcn_wmma_f32_16x16x32_bf16(
              false, ahi[mt].v, false, blo[nt].v, (short)0, acc[mt][nt], false, false);
          acc[mt][nt] = __builtin_amdgcn_wmma_f32_16x16x32_bf16(
              false, alo[mt].v, false, bhi[nt].v, (short)0, acc[mt][nt], false, false);
        }
      }
    }
  }

#pragma unroll
  for (int mt = 0; mt < 2; ++mt) {
#pragma unroll
    for (int nt = 0; nt < 4; ++nt) {
#pragma unroll
      for (int r = 0; r < 8; ++r) {
        int co = co0 + waveM * 32 + mt * 16 + r + (laneHi ? 8 : 0);
        int w  = w0 + waveN * 64 + nt * 16 + lm;
        out[((size_t)b * C_OUT + co) * (size_t)W_IN + w] = acc[mt][nt][r] + bias[co];
      }
    }
  }
}

extern "C" void kernel_launch(void* const* d_in, const int* in_sizes, int n_in,
                              void* d_out, int out_size, void* d_ws, size_t ws_size,
                              hipStream_t stream) {
  const float* in   = (const float*)d_in[0];
  const float* wt   = (const float*)d_in[1];
  const float* bias = (const float*)d_in[2];
  float* out = (float*)d_out;

  const size_t need = (2 * SI_ELEMS + 2 * SW_ELEMS) * sizeof(unsigned short);
  dim3 grid(W_IN / NTILE, C_OUT / MTILE, BATCH);   // 32 x 2 x 32

  if (ws_size >= need && d_ws != nullptr) {
    unsigned short* wIhi = (unsigned short*)d_ws;
    unsigned short* wIlo = wIhi + SI_ELEMS;
    unsigned short* wWhi = wIlo + SI_ELEMS;
    unsigned short* wWlo = wWhi + SW_ELEMS;
    prep_input<<<dim3((WPADD + 255) / 256, NCHUNK, BATCH), 256, 0, stream>>>(in, wIhi, wIlo);
    prep_weight<<<dim3((unsigned)((SW_ELEMS + 255) / 256)), 256, 0, stream>>>(wt, wWhi, wWlo);
    conv1d_wmma_bf16x3_async<<<grid, 256, 0, stream>>>(wIhi, wIlo, wWhi, wWlo, bias, out);
  } else {
    conv1d_wmma_bf16x3_fb<<<grid, 256, 0, stream>>>(in, wt, bias, out);
  }
}
